// MinGRUCUDA_77309411813
// MI455X (gfx1250) — matrix operand order repaired and checked by reading, not compiled, not run on hardware
//
#include <hip/hip_runtime.h>
#include <hip/hip_bf16.h>
#include <cstdint>
#include <cstddef>

// MinGRU: gates = sigmoid(x @ Wg^T + bg), cands = tanh(x @ Wc^T + bc)
// h_t = (1-g_t) h_{t-1} + g_t c_t  along L, h_0 = 0.
// B=4, L=8192, D=512.

typedef __attribute__((ext_vector_type(16))) __bf16 v16bf;
typedef __attribute__((ext_vector_type(8)))  float  v8f;
typedef __attribute__((ext_vector_type(4)))  __bf16 v4bf;

#define B_      4
#define L_      8192
#define D_      512
#define M_TOT   (B_ * L_)      // 32768 rows of X
#define TM      128            // block tile M (= time-chunk length)
#define TN      64             // block tile N (channels)
#define TK      32             // K step (bf16 WMMA K)
#define NCHUNK  (L_ / TM)      // 64 chunks per (batch, channel)

// ---------------------------------------------------------------------------
// Kernel 1: tiled GEMM (both Wg and Wc vs the same X tile) via
// v_wmma_f32_16x16x32_bf16, fused with bias+activations and the chunk-local
// scan.  Writes locally-scanned h into `out`, per-element prefix products of
// a=(1-g) into `prefixA`, and per-chunk affine aggregates (A,B).
// Block: 256 threads = 8 waves laid out 4(M) x 2(N); each wave owns a 32x32
// output region = 2x2 WMMA 16x16 subtiles, with two accumulator sets (g, c).
// ---------------------------------------------------------------------------
__global__ __launch_bounds__(256)
void mingru_gemm_scan_kernel(const float* __restrict__ x,
                             const float* __restrict__ Wg,
                             const float* __restrict__ bg,
                             const float* __restrict__ Wc,
                             const float* __restrict__ bc,
                             float* __restrict__ out,       // locally-scanned h
                             float* __restrict__ prefixA,   // prefix prod of a
                             float* __restrict__ Achunk,    // per-chunk prod(a)
                             float* __restrict__ Bchunk)    // per-chunk local h_end
{
    // 64 KB LDS, time-multiplexed:
    //   GEMM staging: ldsX[128][32] bf16 (8KB) | ldsWg[64][32] (4KB) | ldsWc[64][32] (4KB)
    //   scan buffers: sA[128][64] f32 (32KB)   | sB[128][64] f32 (32KB)
    __shared__ float smemf[16384];

    const int tid    = threadIdx.x;
    const int mt     = blockIdx.x;          // 0..255  (M tiles)
    const int nt     = blockIdx.y;          // 0..7    (N tiles)
    const int m0     = mt * TM;
    const int n0     = nt * TN;

    const int lane   = tid & 31;
    const int wave   = tid >> 5;            // 0..7
    const int wm     = wave & 3;            // M wave group
    const int wn     = wave >> 2;           // N wave group
    const int lanelo = lane & 15;
    const int laneHi = lane >> 4;           // 0 or 1

    __bf16* ldsX  = (__bf16*)smemf;                 // [128][32]
    __bf16* ldsWg = (__bf16*)(smemf + 2048);        // [64][32] (byte off 8KB)
    __bf16* ldsWc = (__bf16*)(smemf + 3072);        // [64][32] (byte off 12KB)
    float*  sA    = smemf;                          // [128][64]
    float*  sB    = smemf + 8192;                   // [128][64]

    v8f accG[2][2] = {};
    v8f accC[2][2] = {};

    const int rgrp = tid >> 3;          // 0..31 rows per staging pass
    const int cg   = (tid & 7) * 4;     // 0,4,...,28 column group (float4)

    for (int kt = 0; kt < D_ / TK; ++kt) {
        const int k0 = kt * TK;
        __syncthreads();  // protect staging LDS from previous iteration reads

        // --- stage X tile: 128 rows x 32 cols fp32 -> bf16 -----------------
        #pragma unroll
        for (int p = 0; p < 4; ++p) {
            const int row = p * 32 + rgrp;
            const float4 v = *(const float4*)(x + (size_t)(m0 + row) * D_ + k0 + cg);
            v4bf bv;
            bv[0] = (__bf16)v.x; bv[1] = (__bf16)v.y;
            bv[2] = (__bf16)v.z; bv[3] = (__bf16)v.w;
            *(v4bf*)(ldsX + row * 32 + cg) = bv;
        }
        // --- stage Wg / Wc tiles: 64 rows (channels) x 32 cols (k) ---------
        // B[k][n] = W[e=n0+n][d=k]; store row-major by channel, read along k.
        #pragma unroll
        for (int p = 0; p < 2; ++p) {
            const int row = p * 32 + rgrp;  // 0..63
            const float4 vg = *(const float4*)(Wg + (size_t)(n0 + row) * D_ + k0 + cg);
            const float4 vc = *(const float4*)(Wc + (size_t)(n0 + row) * D_ + k0 + cg);
            v4bf g4, c4;
            g4[0] = (__bf16)vg.x; g4[1] = (__bf16)vg.y;
            g4[2] = (__bf16)vg.z; g4[3] = (__bf16)vg.w;
            c4[0] = (__bf16)vc.x; c4[1] = (__bf16)vc.y;
            c4[2] = (__bf16)vc.z; c4[3] = (__bf16)vc.w;
            *(v4bf*)(ldsWg + row * 32 + cg) = g4;
            *(v4bf*)(ldsWc + row * 32 + cg) = c4;
        }
        __syncthreads();

        // --- build WMMA fragments per documented gfx1250 layouts -----------
        // A (16x32 bf16): lanes 0-15 -> M=lane, k in {0..7,16..23};
        //                 lanes 16-31 -> M=lane-16, k in {8..15,24..31}.
        union Frag { v16bf v; uint4 q[2]; };
        Frag aF[2];
        #pragma unroll
        for (int i = 0; i < 2; ++i) {
            const int row = wm * 32 + i * 16 + lanelo;
            const __bf16* p = ldsX + row * 32 + 8 * laneHi;
            aF[i].q[0] = *(const uint4*)p;         // k = 8*hi + 0..7
            aF[i].q[1] = *(const uint4*)(p + 16);  // k = 16 + 8*hi + 0..7
        }
        // B (32x16 bf16): lanes 0-15 -> N=lane, k in {0..7,16..23};
        //                 lanes 16-31 -> N=lane-16, k in {8..15,24..31}.
        Frag bGF[2], bCF[2];
        #pragma unroll
        for (int j = 0; j < 2; ++j) {
            const int nn = wn * 32 + j * 16 + lanelo;
            const __bf16* pg = ldsWg + nn * 32 + 8 * laneHi;
            bGF[j].q[0] = *(const uint4*)pg;
            bGF[j].q[1] = *(const uint4*)(pg + 16);
            const __bf16* pc = ldsWc + nn * 32 + 8 * laneHi;
            bCF[j].q[0] = *(const uint4*)pc;
            bCF[j].q[1] = *(const uint4*)(pc + 16);
        }
        // --- 8 WMMAs per K-step per wave -----------------------------------
        #pragma unroll
        for (int i = 0; i < 2; ++i) {
            #pragma unroll
            for (int j = 0; j < 2; ++j) {
                accG[i][j] = __builtin_amdgcn_wmma_f32_16x16x32_bf16(
                    false, aF[i].v, false, bGF[j].v, (short)0, accG[i][j], false, false);
                accC[i][j] = __builtin_amdgcn_wmma_f32_16x16x32_bf16(
                    false, aF[i].v, false, bCF[j].v, (short)0, accC[i][j], false, false);
            }
        }
    }
    __syncthreads();  // staging LDS done; switch to scan buffers

    // --- epilogue: bias + activations, form a = 1-g and b = g*c in LDS -----
    // C/D layout: VGPR r, lanes 0-15 -> (M=r, N=lane); lanes 16-31 -> (M=r+8).
    float bgv[2], bcv[2];
    #pragma unroll
    for (int j = 0; j < 2; ++j) {
        const int n = n0 + wn * 32 + j * 16 + lanelo;
        bgv[j] = bg[n];
        bcv[j] = bc[n];
    }
    #pragma unroll
    for (int i = 0; i < 2; ++i) {
        #pragma unroll
        for (int j = 0; j < 2; ++j) {
            #pragma unroll
            for (int r = 0; r < 8; ++r) {
                const int ml = wm * 32 + i * 16 + r + 8 * laneHi;
                const int nl = wn * 32 + j * 16 + lanelo;
                const float g = 1.0f / (1.0f + __expf(-(accG[i][j][r] + bgv[j])));
                const float c = tanhf(accC[i][j][r] + bcv[j]);
                sA[ml * 64 + nl] = 1.0f - g;
                sB[ml * 64 + nl] = g * c;
            }
        }
    }
    __syncthreads();

    // --- chunk-local scan: one thread per channel, 128 sequential steps ----
    if (tid < TN) {
        const int e     = n0 + tid;
        const int b     = m0 / L_;
        const int l0    = m0 % L_;
        const int chunk = l0 / TM;
        float h  = 0.0f;   // local scan assuming h_in = 0
        float pA = 1.0f;   // running product of a (for carry fix-up)
        for (int t = 0; t < TM; ++t) {
            const float a  = sA[t * 64 + tid];
            const float bb = sB[t * 64 + tid];
            h  = fmaf(a, h, bb);
            pA *= a;
            const size_t gi = (size_t)(m0 + t) * D_ + e;
            out[gi]     = h;   // h_local
            prefixA[gi] = pA;  // prod_{s<=t} a_s
        }
        const size_t ci = ((size_t)b * NCHUNK + chunk) * D_ + e;
        Achunk[ci] = pA;  // chunk transfer: h_out = A*h_in + B
        Bchunk[ci] = h;
    }
}

// ---------------------------------------------------------------------------
// Kernel 2: sequential scan over the 64 chunk aggregates per (b, e).
// carry[b][k][e] = h entering chunk k.  2048 independent, coalesced threads.
// ---------------------------------------------------------------------------
__global__ __launch_bounds__(256)
void mingru_carry_kernel(const float* __restrict__ Achunk,
                         const float* __restrict__ Bchunk,
                         float* __restrict__ carry)
{
    const int gid = blockIdx.x * blockDim.x + threadIdx.x;
    if (gid >= B_ * D_) return;
    const int b = gid / D_;
    const int e = gid % D_;
    float H = 0.0f;
    for (int k = 0; k < NCHUNK; ++k) {
        const size_t ci = ((size_t)b * NCHUNK + k) * D_ + e;
        carry[ci] = H;
        H = fmaf(Achunk[ci], H, Bchunk[ci]);
    }
}

// ---------------------------------------------------------------------------
// Kernel 3: out = h_local + prefixA * carry[chunk], float4-vectorized.
// ---------------------------------------------------------------------------
__global__ __launch_bounds__(256)
void mingru_apply_kernel(const float* __restrict__ prefixA,
                         const float* __restrict__ carry,
                         float* __restrict__ out)
{
    const size_t gid4 = (size_t)blockIdx.x * blockDim.x + threadIdx.x;  // BLD/4 total
    const int ncol4 = D_ / 4;
    const int e4    = (int)(gid4 % ncol4);
    const int m     = (int)(gid4 / ncol4);
    const int b     = m / L_;
    const int l     = m % L_;
    const int chunk = l / TM;
    const size_t gi = (size_t)m * D_ + e4 * 4;
    const size_t ci = ((size_t)b * NCHUNK + chunk) * D_ + e4 * 4;

    const float4 pa = *(const float4*)(prefixA + gi);
    const float4 cr = *(const float4*)(carry + ci);
    float4 o        = *(float4*)(out + gi);
    o.x = fmaf(pa.x, cr.x, o.x);
    o.y = fmaf(pa.y, cr.y, o.y);
    o.z = fmaf(pa.z, cr.z, o.z);
    o.w = fmaf(pa.w, cr.w, o.w);
    *(float4*)(out + gi) = o;
}

// ---------------------------------------------------------------------------
extern "C" void kernel_launch(void* const* d_in, const int* in_sizes, int n_in,
                              void* d_out, int out_size, void* d_ws, size_t ws_size,
                              hipStream_t stream)
{
    (void)in_sizes; (void)n_in; (void)out_size; (void)ws_size;

    const float* x  = (const float*)d_in[0];
    const float* Wg = (const float*)d_in[1];
    const float* bg = (const float*)d_in[2];
    const float* Wc = (const float*)d_in[3];
    const float* bc = (const float*)d_in[4];
    float* out = (float*)d_out;

    // Workspace layout (floats):
    //   prefixA : B*L*D                 (64 MB)
    //   Achunk  : B*NCHUNK*D            (0.5 MB)
    //   Bchunk  : B*NCHUNK*D            (0.5 MB)
    //   carry   : B*NCHUNK*D            (0.5 MB)
    float* prefixA = (float*)d_ws;
    float* Achunk  = prefixA + (size_t)B_ * L_ * D_;
    float* Bchunk  = Achunk  + (size_t)B_ * NCHUNK * D_;
    float* carry   = Bchunk  + (size_t)B_ * NCHUNK * D_;

    dim3 g1(M_TOT / TM, D_ / TN);  // 256 x 8 blocks
    mingru_gemm_scan_kernel<<<g1, 256, 0, stream>>>(x, Wg, bg, Wc, bc,
                                                    out, prefixA, Achunk, Bchunk);

    mingru_carry_kernel<<<(B_ * D_ + 255) / 256, 256, 0, stream>>>(Achunk, Bchunk, carry);

    const int total4 = (B_ * L_ * D_) / 4;
    mingru_apply_kernel<<<(total4 + 255) / 256, 256, 0, stream>>>(prefixA, carry, out);
}
